// LlamaAttention_18846316495171
// MI455X (gfx1250) — matrix operand order repaired and checked by reading, not compiled, not run on hardware
//
#include <hip/hip_runtime.h>

typedef __attribute__((ext_vector_type(16))) __bf16 bf16x16;
typedef __attribute__((ext_vector_type(8)))  __bf16 bf16x8;
typedef __attribute__((ext_vector_type(4)))  __bf16 bf16x4;
typedef __attribute__((ext_vector_type(8)))  float  f32x8;
typedef __attribute__((ext_vector_type(4)))  float  f32x4;
typedef __attribute__((ext_vector_type(4)))  int    i32x4;

constexpr int S  = 2048;
constexpr int H  = 4096;
constexpr int NH = 32;
constexpr int HD = 128;
constexpr int DQ = NH * HD;   // 4096

#if defined(__AMDGCN__) && __has_builtin(__builtin_amdgcn_global_load_async_to_lds_b128)
#define USE_ASYNC_COPY 1
#define ASYNC_GPTR(p) ((__attribute__((address_space(1))) i32x4*)(p))
#define ASYNC_SPTR(p) ((__attribute__((address_space(3))) i32x4*)(p))
#else
#define USE_ASYNC_COPY 0
#endif

__device__ __forceinline__ void wait_async0() {
#if defined(__AMDGCN__)
#if __has_builtin(__builtin_amdgcn_s_wait_asynccnt)
  __builtin_amdgcn_s_wait_asynccnt(0);
#else
  asm volatile("s_wait_asynccnt 0x0" ::: "memory");
#endif
#endif
}

__device__ __forceinline__ bf16x16 cat8(bf16x8 a, bf16x8 b) {
  return __builtin_shufflevector(a, b, 0,1,2,3,4,5,6,7,8,9,10,11,12,13,14,15);
}
__device__ __forceinline__ f32x8 wmma_bf16(bf16x16 a, bf16x16 b, f32x8 c) {
  return __builtin_amdgcn_wmma_f32_16x16x32_bf16(false, a, false, b, (short)0, c,
                                                 false, false);
}

// ---------------------------------------------------------------------------
// GEMM: C[M,N] = A[M,K] * B[N,K]^T   (contraction over K = weight row dim)
// 128x128 block tile, 8 waves; each wave 32(M) x 64(N) -> 2x4 accumulators.
// Double-buffered LDS; vectorized staging with packed f32->bf16 conversion.
// ---------------------------------------------------------------------------
template <typename AT, typename OT>
__global__ __launch_bounds__(256) void gemm_xwT(const AT* __restrict__ A,
                                                const float* __restrict__ B,
                                                OT* __restrict__ C,
                                                int M, int N, int K) {
  __shared__ __bf16 As[2][128][32];
  __shared__ __bf16 Bs[2][128][32];

  const int t    = threadIdx.x;
  const int lane = t & 31;
  const int wave = t >> 5;
  const int half = lane >> 4;
  const int l16  = lane & 15;
  const int wm   = wave & 3;   // 4 sub-tiles of 32 rows
  const int wn   = wave >> 2;  // 2 sub-tiles of 64 cols
  const int m0   = blockIdx.y * 128;
  const int n0   = blockIdx.x * 128;

  f32x4  aR[4];
  bf16x8 aR8[2];
  f32x4  bR[4];

  auto load_tiles = [&](int kk) {
    if constexpr (sizeof(AT) == 4) {
#pragma unroll
      for (int i = 0; i < 4; ++i) {
        int v = i * 256 + t, r = v >> 3, cg = (v & 7) * 4;
        aR[i] = *(const f32x4*)&A[(size_t)(m0 + r) * K + kk + cg];
      }
    } else {
#pragma unroll
      for (int i = 0; i < 2; ++i) {
        int v = i * 256 + t, r = v >> 2, cg = (v & 3) * 8;
        aR8[i] = *(const bf16x8*)&A[(size_t)(m0 + r) * K + kk + cg];
      }
    }
#pragma unroll
    for (int i = 0; i < 4; ++i) {
      int v = i * 256 + t, r = v >> 3, cg = (v & 7) * 4;
      bR[i] = *(const f32x4*)&B[(size_t)(n0 + r) * K + kk + cg];
    }
  };
  auto store_tiles = [&](int buf) {
    if constexpr (sizeof(AT) == 4) {
#pragma unroll
      for (int i = 0; i < 4; ++i) {
        int v = i * 256 + t, r = v >> 3, cg = (v & 7) * 4;
        *(bf16x4*)&As[buf][r][cg] = __builtin_convertvector(aR[i], bf16x4);
      }
    } else {
#pragma unroll
      for (int i = 0; i < 2; ++i) {
        int v = i * 256 + t, r = v >> 2, cg = (v & 3) * 8;
        *(bf16x8*)&As[buf][r][cg] = aR8[i];
      }
    }
#pragma unroll
    for (int i = 0; i < 4; ++i) {
      int v = i * 256 + t, r = v >> 3, cg = (v & 7) * 4;
      *(bf16x4*)&Bs[buf][r][cg] = __builtin_convertvector(bR[i], bf16x4);
    }
  };

  const f32x8 z = {0.f,0.f,0.f,0.f,0.f,0.f,0.f,0.f};
  f32x8 acc[2][4];
#pragma unroll
  for (int mt = 0; mt < 2; ++mt)
#pragma unroll
    for (int nt = 0; nt < 4; ++nt) acc[mt][nt] = z;

  load_tiles(0);
  store_tiles(0);
  __syncthreads();

  int cur = 0;
  for (int k0 = 0; k0 < K; k0 += 32) {
    const bool nxt = (k0 + 32) < K;
    if (nxt) load_tiles(k0 + 32);  // global loads overlap WMMA below

    bf16x16 af[2], bf[4];
#pragma unroll
    for (int mt = 0; mt < 2; ++mt) {
      const __bf16* p = &As[cur][wm * 32 + mt * 16 + l16][0];
      af[mt] = cat8(*(const bf16x8*)(p + 8 * half),
                    *(const bf16x8*)(p + 16 + 8 * half));
    }
#pragma unroll
    for (int nt = 0; nt < 4; ++nt) {
      const __bf16* p = &Bs[cur][wn * 64 + nt * 16 + l16][0];
      bf[nt] = cat8(*(const bf16x8*)(p + 8 * half),
                    *(const bf16x8*)(p + 16 + 8 * half));
    }
#pragma unroll
    for (int mt = 0; mt < 2; ++mt)
#pragma unroll
      for (int nt = 0; nt < 4; ++nt)
        acc[mt][nt] = wmma_bf16(af[mt], bf[nt], acc[mt][nt]);

    if (nxt) store_tiles(cur ^ 1);
    __syncthreads();
    cur ^= 1;
  }

#pragma unroll
  for (int mt = 0; mt < 2; ++mt)
#pragma unroll
    for (int nt = 0; nt < 4; ++nt)
#pragma unroll
      for (int r = 0; r < 8; ++r) {
        int row = m0 + wm * 32 + mt * 16 + 8 * half + r;
        int col = n0 + wn * 64 + nt * 16 + l16;
        C[(size_t)row * N + col] = (OT)acc[mt][nt][r];
      }
}

// ---------------------------------------------------------------------------
// RoPE applied in place to bf16 Q and K laid out [S][NH*HD].
// ---------------------------------------------------------------------------
__global__ void rope_qk(__bf16* __restrict__ Q, __bf16* __restrict__ K,
                        const int* __restrict__ pos) {
  int idx = blockIdx.x * blockDim.x + threadIdx.x;
  if (idx >= S * NH * (HD / 2)) return;
  int d = idx & 63;
  int h = (idx >> 6) & 31;
  int s = idx >> 11;
  float p   = (float)pos[s];
  float inv = __expf(-9.210340371976184f * ((float)(2 * d) * (1.0f / 128.0f)));
  float th  = p * inv;
  float sn, cs;
  __sincosf(th, &sn, &cs);
  size_t base = (size_t)s * DQ + h * HD + d;
  float q1 = (float)Q[base], q2 = (float)Q[base + 64];
  Q[base]      = (__bf16)(q1 * cs - q2 * sn);
  Q[base + 64] = (__bf16)(q2 * cs + q1 * sn);
  float k1 = (float)K[base], k2 = (float)K[base + 64];
  K[base]      = (__bf16)(k1 * cs - k2 * sn);
  K[base + 64] = (__bf16)(k2 * cs + k1 * sn);
}

// ---------------------------------------------------------------------------
// Flash attention, causal. One block per (head, 128-query rows); each wave
// owns 16 query rows; key blocks of 32. K staged once per block into LDS
// (async global->LDS when available), V staged transposed for P*V B-frags.
// ---------------------------------------------------------------------------
__global__ __launch_bounds__(256) void flash_attn(const __bf16* __restrict__ Q,
                                                  const __bf16* __restrict__ K,
                                                  const __bf16* __restrict__ V,
                                                  __bf16* __restrict__ O) {
  __shared__ __bf16 Kl[32][128];     // K rows as-is: [key][hd]
  __shared__ __bf16 Vt[128][32];     // V transposed: [hd][key]
  __shared__ __bf16 Pl[8][16][32];   // per-wave P tile (C-layout -> A-layout)

  const int t    = threadIdx.x;
  const int lane = t & 31;
  const int w    = t >> 5;
  const int half = lane >> 4;
  const int l16  = lane & 15;
  const int hd0   = blockIdx.x * HD;
  const int qbase = blockIdx.y * 128;
  const int qrow  = qbase + w * 16 + l16;

  bf16x16 qf[4];
#pragma unroll
  for (int c = 0; c < 4; ++c) {
    const __bf16* p = Q + (size_t)qrow * DQ + hd0 + c * 32;
    qf[c] = cat8(*(const bf16x8*)(p + 8 * half),
                 *(const bf16x8*)(p + 16 + 8 * half));
  }

  const f32x8 z = {0.f,0.f,0.f,0.f,0.f,0.f,0.f,0.f};
  f32x8 o[8];
  float mr[8], lr[8];
#pragma unroll
  for (int nt = 0; nt < 8; ++nt) o[nt] = z;
#pragma unroll
  for (int r = 0; r < 8; ++r) { mr[r] = -3.0e38f; lr[r] = 0.f; }

  const float scale = 0.08838834764831845f;  // 1/sqrt(128)
  const int kend = qbase + 128;

  for (int kb = 0; kb < kend; kb += 32) {
    __syncthreads();
    // ---- stage K block straight into LDS (async path if available) ----
#pragma unroll
    for (int i = 0; i < 2; ++i) {
      int g = i * 256 + t;
      int key = g >> 4, hg = (g & 15) * 8;
#if USE_ASYNC_COPY
      __builtin_amdgcn_global_load_async_to_lds_b128(
          ASYNC_GPTR(K + (size_t)(kb + key) * DQ + hd0 + hg),
          ASYNC_SPTR(&Kl[key][hg]), 0, 0);
#else
      *(bf16x8*)&Kl[key][hg] =
          *(const bf16x8*)(K + (size_t)(kb + key) * DQ + hd0 + hg);
#endif
    }
    // ---- stage V block transposed into LDS ----
#pragma unroll
    for (int i = 0; i < 2; ++i) {
      int g = i * 256 + t;
      int key = g >> 4, hg = (g & 15) * 8;
      bf16x8 v = *(const bf16x8*)(V + (size_t)(kb + key) * DQ + hd0 + hg);
#pragma unroll
      for (int j = 0; j < 8; ++j) Vt[hg + j][key] = v[j];
    }
#if USE_ASYNC_COPY
    wait_async0();
#endif
    __syncthreads();

    // ---- scores S = Q * K^T, two 16-key tiles, K-frags from LDS ----
    f32x8 sc[2] = {z, z};
#pragma unroll
    for (int kt = 0; kt < 2; ++kt) {
      const __bf16* kp = &Kl[kt * 16 + l16][0];
#pragma unroll
      for (int c = 0; c < 4; ++c) {
        const __bf16* p = kp + c * 32;
        bf16x16 bfr = cat8(*(const bf16x8*)(p + 8 * half),
                           *(const bf16x8*)(p + 16 + 8 * half));
        sc[kt] = wmma_bf16(qf[c], bfr, sc[kt]);
      }
    }

    // ---- online softmax ----
#pragma unroll
    for (int r = 0; r < 8; ++r) {
      int qg = qbase + w * 16 + 8 * half + r;
      float s0 = sc[0][r] * scale;
      if (kb + l16 > qg) s0 = -3.0e38f;
      float s1 = sc[1][r] * scale;
      if (kb + 16 + l16 > qg) s1 = -3.0e38f;
      float mx = fmaxf(s0, s1);
      mx = fmaxf(mx, __shfl_xor(mx, 1, 32));
      mx = fmaxf(mx, __shfl_xor(mx, 2, 32));
      mx = fmaxf(mx, __shfl_xor(mx, 4, 32));
      mx = fmaxf(mx, __shfl_xor(mx, 8, 32));
      float mn = fmaxf(mr[r], mx);
      float al = __expf(mr[r] - mn);
      float p0 = __expf(s0 - mn);
      float p1 = __expf(s1 - mn);
      float rs = p0 + p1;
      rs += __shfl_xor(rs, 1, 32);
      rs += __shfl_xor(rs, 2, 32);
      rs += __shfl_xor(rs, 4, 32);
      rs += __shfl_xor(rs, 8, 32);
      lr[r] = lr[r] * al + rs;
      mr[r] = mn;
#pragma unroll
      for (int nt = 0; nt < 8; ++nt) o[nt][r] *= al;
      Pl[w][8 * half + r][l16]      = (__bf16)p0;
      Pl[w][8 * half + r][16 + l16] = (__bf16)p1;
    }
    __syncthreads();

    // ---- O += P(16x32) * V(32x128) ----
    bf16x16 pf;
    {
      const __bf16* p = &Pl[w][l16][0];
      pf = cat8(*(const bf16x8*)(p + 8 * half),
                *(const bf16x8*)(p + 16 + 8 * half));
    }
#pragma unroll
    for (int nt = 0; nt < 8; ++nt) {
      const __bf16* vp = &Vt[nt * 16 + l16][0];
      bf16x16 bfr = cat8(*(const bf16x8*)(vp + 8 * half),
                         *(const bf16x8*)(vp + 16 + 8 * half));
      o[nt] = wmma_bf16(pf, bfr, o[nt]);
    }
  }

#pragma unroll
  for (int nt = 0; nt < 8; ++nt)
#pragma unroll
    for (int r = 0; r < 8; ++r) {
      int qg = qbase + w * 16 + 8 * half + r;
      float val = o[nt][r] / lr[r];
      O[(size_t)qg * DQ + hd0 + nt * 16 + l16] = (__bf16)val;
    }
}

// ---------------------------------------------------------------------------
// inputs: 0 hidden_states f32, 1 attention_mask (causal, computed
// analytically), 2 position_ids i32, 3 Wq, 4 Wk, 5 Wv, 6 Wo (all f32)
// ---------------------------------------------------------------------------
extern "C" void kernel_launch(void* const* d_in, const int* in_sizes, int n_in,
                              void* d_out, int out_size, void* d_ws,
                              size_t ws_size, hipStream_t stream) {
  (void)in_sizes; (void)n_in; (void)out_size; (void)ws_size;
  const float* X   = (const float*)d_in[0];
  const int*   pos = (const int*)d_in[2];
  const float* Wq  = (const float*)d_in[3];
  const float* Wk  = (const float*)d_in[4];
  const float* Wv  = (const float*)d_in[5];
  const float* Wo  = (const float*)d_in[6];
  float* out = (float*)d_out;

  const size_t n = (size_t)S * DQ;
  __bf16* Qb = (__bf16*)d_ws;
  __bf16* Kb = Qb + n;
  __bf16* Vb = Kb + n;
  __bf16* Ab = Vb + n;

  dim3 g(DQ / 128, S / 128);  // (32, 16)
  gemm_xwT<float, __bf16><<<g, 256, 0, stream>>>(X, Wq, Qb, S, DQ, H);
  gemm_xwT<float, __bf16><<<g, 256, 0, stream>>>(X, Wk, Kb, S, DQ, H);
  gemm_xwT<float, __bf16><<<g, 256, 0, stream>>>(X, Wv, Vb, S, DQ, H);

  int total = S * NH * (HD / 2);
  rope_qk<<<(total + 255) / 256, 256, 0, stream>>>(Qb, Kb, pos);

  flash_attn<<<dim3(NH, S / 128), 256, 0, stream>>>(Qb, Kb, Vb, Ab);

  gemm_xwT<__bf16, float><<<dim3(H / 128, S / 128), 256, 0, stream>>>(
      Ab, Wo, out, S, H, DQ);
}
